// DDiTBlock_86912958202367
// MI455X (gfx1250) — matrix operand order repaired and checked by reading, not compiled, not run on hardware
//
#include <hip/hip_runtime.h>
#include <stdint.h>

#define DIM   1024
#define NH    16
#define HD    64
#define B_    8
#define L_    1024
#define T_    (B_ * L_)
#define MLPD  4096
#define ADAD  (6 * DIM)
#define EPSV  1e-5f

typedef __bf16 bf16_t;
typedef __attribute__((ext_vector_type(16))) __bf16 v16bf;
typedef __attribute__((ext_vector_type(8)))  float  v8f;

static __device__ inline bf16_t f2bf(float f) {
  union { float f; uint32_t u; } v; v.f = f;
  uint32_t r = v.u + 0x7fffu + ((v.u >> 16) & 1u);
  uint16_t h = (uint16_t)(r >> 16);
  bf16_t o; __builtin_memcpy(&o, &h, 2); return o;
}

static __device__ inline v16bf frag_from(const bf16_t* p0, const bf16_t* p1) {
  v16bf r;
#pragma unroll
  for (int i = 0; i < 8; ++i) { r[i] = p0[i]; r[i + 8] = p1[i]; }
  return r;
}

static __device__ inline float gelu_tanh(float x) {
  float x3 = x * x * x;
  return 0.5f * x * (1.f + tanhf(0.7978845608028654f * (x + 0.044715f * x3)));
}

// ---------------------------------------------------------------------------
// Weight prep: fp32 [R][C] -> bf16 transposed [C][R]
// ---------------------------------------------------------------------------
__global__ void k_transpose_bf16(const float* __restrict__ src,
                                 bf16_t* __restrict__ dst, int R, int C) {
  int idx = blockIdx.x * 256 + threadIdx.x;
  if (idx >= R * C) return;
  int r = idx / C, c = idx % C;
  dst[(size_t)c * R + r] = f2bf(src[(size_t)r * C + c]);
}

// ---------------------------------------------------------------------------
// ada = c @ ada_w + ada_b  (tiny: 8 x 6144)
// ---------------------------------------------------------------------------
__global__ void k_ada(const float* __restrict__ c, const float* __restrict__ w,
                      const float* __restrict__ b, float* __restrict__ out) {
  int idx = blockIdx.x * 256 + threadIdx.x;   // B_*ADAD exact
  int bb = idx / ADAD, j = idx % ADAD;
  float acc = b[j];
  const float* cr = c + (size_t)bb * DIM;
  for (int k = 0; k < DIM; ++k) acc += cr[k] * w[(size_t)k * ADAD + j];
  out[idx] = acc;
}

// ---------------------------------------------------------------------------
// h = bf16( LN(x) * (1 + ada[sc]) + ada[sh] )   one block per token
// ---------------------------------------------------------------------------
__global__ __launch_bounds__(256)
void k_ln_mod(const float* __restrict__ x, const float* __restrict__ lnw,
              const float* __restrict__ ada, int sh_off, int sc_off,
              bf16_t* __restrict__ out) {
  const int t = blockIdx.x, tid = threadIdx.x;
  __shared__ float red[256];
  const float* row = x + (size_t)t * DIM;
  float vals[4]; float s = 0.f;
#pragma unroll
  for (int i = 0; i < 4; ++i) { vals[i] = row[tid + 256 * i]; s += vals[i]; }
  red[tid] = s; __syncthreads();
  for (int o = 128; o > 0; o >>= 1) { if (tid < o) red[tid] += red[tid + o]; __syncthreads(); }
  const float mu = red[0] * (1.f / DIM);
  __syncthreads();
  float sq = 0.f;
#pragma unroll
  for (int i = 0; i < 4; ++i) { float d = vals[i] - mu; sq += d * d; }
  red[tid] = sq; __syncthreads();
  for (int o = 128; o > 0; o >>= 1) { if (tid < o) red[tid] += red[tid + o]; __syncthreads(); }
  const float rstd = rsqrtf(red[0] * (1.f / DIM) + EPSV);
  const int seq = t / L_;
  const float* ab = ada + (size_t)seq * ADAD;
#pragma unroll
  for (int i = 0; i < 4; ++i) {
    int dcol = tid + 256 * i;
    float h = (vals[i] - mu) * rstd * lnw[dcol];
    h = h * (1.f + ab[sc_off + dcol]) + ab[sh_off + dcol];
    out[(size_t)t * DIM + dcol] = f2bf(h);
  }
}

// ---------------------------------------------------------------------------
// WMMA GEMM: C[M,N] = A[M,K] @ B[K,N], A bf16 row-major, BT bf16 [N][K].
// Block tile 128x256, 8 waves (2x4), wave tile 64x64 (16 WMMA per K-step).
// Ping-pong LDS buffers with register staging: one barrier per K-step,
// global loads for tile i+1 overlap WMMA compute on tile i.
// MODE 0: fp32 out, MODE 2: bias+gelu -> bf16 out.
// ---------------------------------------------------------------------------
#define BM 128
#define BN 256
#define BK 32
#define LPAD 8

template <int MODE>
__global__ __launch_bounds__(256)
void k_gemm(int M, int N, int K,
            const bf16_t* __restrict__ A, const bf16_t* __restrict__ BT,
            float* __restrict__ outF, bf16_t* __restrict__ outB,
            const float* __restrict__ bias) {
  __shared__ bf16_t As[2][BM][BK + LPAD];
  __shared__ bf16_t Bs[2][BN][BK + LPAD];
  const int tid = threadIdx.x;
  const int lane = tid & 31, w = tid >> 5;
  const int wm = w >> 2, wn = w & 3;           // 2 x 4 wave grid
  const int hf = lane >> 4, l16 = lane & 15;
  const int m0 = blockIdx.x * BM;
  const int n0 = blockIdx.y * BN;

  v8f acc[4][4];
#pragma unroll
  for (int mi = 0; mi < 4; ++mi)
#pragma unroll
    for (int ni = 0; ni < 4; ++ni)
#pragma unroll
      for (int j = 0; j < 8; ++j) acc[mi][ni][j] = 0.f;

  // staging: A tile 128x32 -> thread loads half a row (16 bf16);
  //          B tile 256x32 -> thread loads a full row (32 bf16)
  const int ar = tid >> 1, ac = (tid & 1) * 16;
  uint4 ra[2], rb[4];

  const uint4* gA = reinterpret_cast<const uint4*>(A + (size_t)(m0 + ar) * K + ac);
  const uint4* gB = reinterpret_cast<const uint4*>(BT + (size_t)(n0 + tid) * K);
  const int kstep_u4 = BK / 8;   // uint4 elems per BK along K

  // preload tile 0
  ra[0] = gA[0]; ra[1] = gA[1];
  rb[0] = gB[0]; rb[1] = gB[1]; rb[2] = gB[2]; rb[3] = gB[3];
  {
    uint4* da = reinterpret_cast<uint4*>(&As[0][ar][ac]);
    da[0] = ra[0]; da[1] = ra[1];
    uint4* db = reinterpret_cast<uint4*>(&Bs[0][tid][0]);
    db[0] = rb[0]; db[1] = rb[1]; db[2] = rb[2]; db[3] = rb[3];
  }
  __syncthreads();

  const int nk = K / BK;
  for (int it = 0; it < nk; ++it) {
    const int cur = it & 1;
    const bool more = (it + 1) < nk;
    if (more) {  // issue next-tile global loads before compute
      const uint4* sa = gA + (size_t)(it + 1) * kstep_u4;
      ra[0] = sa[0]; ra[1] = sa[1];
      const uint4* sb = gB + (size_t)(it + 1) * kstep_u4;
      rb[0] = sb[0]; rb[1] = sb[1]; rb[2] = sb[2]; rb[3] = sb[3];
    }

    v16bf af[4], bfg[4];
#pragma unroll
    for (int mi = 0; mi < 4; ++mi) {
      const bf16_t* rowp = &As[cur][wm * 64 + mi * 16 + l16][0];
      af[mi] = frag_from(rowp + (hf ? 16 : 0), rowp + (hf ? 24 : 8));
    }
#pragma unroll
    for (int ni = 0; ni < 4; ++ni) {
      const bf16_t* colp = &Bs[cur][wn * 64 + ni * 16 + l16][hf ? 16 : 0];
      bfg[ni] = frag_from(colp, colp + 8);
    }
#pragma unroll
    for (int mi = 0; mi < 4; ++mi)
#pragma unroll
      for (int ni = 0; ni < 4; ++ni)
        acc[mi][ni] = __builtin_amdgcn_wmma_f32_16x16x32_bf16(
            false, af[mi], false, bfg[ni], (short)0, acc[mi][ni], false, false);

    if (more) {  // commit staged regs into the other buffer
      uint4* da = reinterpret_cast<uint4*>(&As[cur ^ 1][ar][ac]);
      da[0] = ra[0]; da[1] = ra[1];
      uint4* db = reinterpret_cast<uint4*>(&Bs[cur ^ 1][tid][0]);
      db[0] = rb[0]; db[1] = rb[1]; db[2] = rb[2]; db[3] = rb[3];
    }
    __syncthreads();
  }

#pragma unroll
  for (int mi = 0; mi < 4; ++mi)
#pragma unroll
    for (int ni = 0; ni < 4; ++ni)
#pragma unroll
      for (int r = 0; r < 8; ++r) {
        int row = m0 + wm * 64 + mi * 16 + hf * 8 + r;
        int col = n0 + wn * 64 + ni * 16 + l16;
        float v = acc[mi][ni][r];
        if (MODE == 2) { v += bias[col]; v = gelu_tanh(v); }
        if (MODE == 0) outF[(size_t)row * N + col] = v;
        else           outB[(size_t)row * N + col] = f2bf(v);
      }
}

// ---------------------------------------------------------------------------
// RoPE (half-split) on q/k + bf16 convert of q/k/v
// ---------------------------------------------------------------------------
__global__ void k_rope_cvt(const float* __restrict__ qf, const float* __restrict__ kf,
                           const float* __restrict__ vf,
                           const float* __restrict__ cosT, const float* __restrict__ sinT,
                           const long long* __restrict__ pos,
                           bf16_t* __restrict__ qb, bf16_t* __restrict__ kb,
                           bf16_t* __restrict__ vb) {
  int idx = blockIdx.x * 256 + threadIdx.x;   // T_*NH*32 exact
  int i = idx & 31;
  int h = (idx >> 5) & (NH - 1);
  int t = idx >> 9;
  long long p = pos[t];
  float c = cosT[p * (HD / 2) + i], s = sinT[p * (HD / 2) + i];
  size_t base = (size_t)t * DIM + h * HD;
  float q1 = qf[base + i], q2 = qf[base + 32 + i];
  qb[base + i]      = f2bf(q1 * c - q2 * s);
  qb[base + 32 + i] = f2bf(q2 * c + q1 * s);
  float k1 = kf[base + i], k2 = kf[base + 32 + i];
  kb[base + i]      = f2bf(k1 * c - k2 * s);
  kb[base + 32 + i] = f2bf(k2 * c + k1 * s);
  vb[base + i]      = f2bf(vf[base + i]);
  vb[base + 32 + i] = f2bf(vf[base + 32 + i]);
}

// ---------------------------------------------------------------------------
// Flash-style attention per (b,h, 128-query block). 8 waves x 16 query rows.
// ---------------------------------------------------------------------------
__global__ __launch_bounds__(256)
void k_attn(const bf16_t* __restrict__ qb, const bf16_t* __restrict__ kb,
            const bf16_t* __restrict__ vb, bf16_t* __restrict__ ob) {
  __shared__ bf16_t Ks[128][HD + LPAD];        // K chunk   [key][d]
  __shared__ bf16_t VTs[HD][128 + LPAD];       // V^T chunk [d][key]
  __shared__ bf16_t Ps[8][16][128 + LPAD];     // per-wave P tile
  const int tid = threadIdx.x;
  const int lane = tid & 31, w = tid >> 5;
  const int hf = lane >> 4, l16 = lane & 15;
  const int b = blockIdx.x >> 4, h = blockIdx.x & (NH - 1);
  const int q0 = blockIdx.y * 128;
  const int seqbase = b * L_;

  // preload Q fragments (A layout), D=64 -> two K-steps of 32
  v16bf aq[2];
  {
    const bf16_t* qp = qb + (size_t)(seqbase + q0 + w * 16 + l16) * DIM + h * HD;
#pragma unroll
    for (int kk = 0; kk < 2; ++kk)
      aq[kk] = frag_from(qp + kk * 32 + (hf ? 16 : 0), qp + kk * 32 + (hf ? 24 : 8));
  }

  v8f oacc[4];
#pragma unroll
  for (int ni = 0; ni < 4; ++ni)
#pragma unroll
    for (int j = 0; j < 8; ++j) oacc[ni][j] = 0.f;
  float mrow[8], lrow[8];
#pragma unroll
  for (int r = 0; r < 8; ++r) { mrow[r] = -1e30f; lrow[r] = 0.f; }

  const int lr = tid >> 1, lcb = (tid & 1) * 32;

  for (int j0 = 0; j0 < L_; j0 += 128) {
    { // stage K (direct) and V (transposed) chunks
      int ktok = seqbase + j0 + lr;
      const uint4* s = reinterpret_cast<const uint4*>(kb + (size_t)ktok * DIM + h * HD + lcb);
      uint4* d = reinterpret_cast<uint4*>(&Ks[lr][lcb]);
      d[0] = s[0]; d[1] = s[1]; d[2] = s[2]; d[3] = s[3];
      const bf16_t* vs = vb + (size_t)ktok * DIM + h * HD + lcb;
#pragma unroll
      for (int i = 0; i < 32; ++i) VTs[lcb + i][lr] = vs[i];
    }
    __syncthreads();

    // S = (Q K^T) / sqrt(D)
    v8f sacc[8];
#pragma unroll
    for (int n = 0; n < 8; ++n)
#pragma unroll
      for (int j = 0; j < 8; ++j) sacc[n][j] = 0.f;
#pragma unroll
    for (int kk = 0; kk < 2; ++kk)
#pragma unroll
      for (int n = 0; n < 8; ++n) {
        const bf16_t* p = &Ks[n * 16 + l16][kk * 32 + (hf ? 16 : 0)];
        v16bf bfrag = frag_from(p, p + 8);
        sacc[n] = __builtin_amdgcn_wmma_f32_16x16x32_bf16(
            false, aq[kk], false, bfrag, (short)0, sacc[n], false, false);
      }
    const float sc = 0.125f;  // 1/sqrt(64)
#pragma unroll
    for (int n = 0; n < 8; ++n)
#pragma unroll
      for (int r = 0; r < 8; ++r) sacc[n][r] *= sc;

    // online softmax per row (rows 0..7 in lanes<16, rows 8..15 in lanes>=16)
#pragma unroll
    for (int r = 0; r < 8; ++r) {
      float m = -1e30f;
#pragma unroll
      for (int n = 0; n < 8; ++n) m = fmaxf(m, sacc[n][r]);
      for (int o = 8; o > 0; o >>= 1) m = fmaxf(m, __shfl_xor(m, o, 16));
      float mn = fmaxf(mrow[r], m);
      float scl = __expf(mrow[r] - mn);
      mrow[r] = mn;
#pragma unroll
      for (int ni = 0; ni < 4; ++ni) oacc[ni][r] *= scl;
      float rs = 0.f;
#pragma unroll
      for (int n = 0; n < 8; ++n) {
        float pv = __expf(sacc[n][r] - mn);
        sacc[n][r] = pv;
        rs += pv;
      }
      for (int o = 8; o > 0; o >>= 1) rs += __shfl_xor(rs, o, 16);
      lrow[r] = lrow[r] * scl + rs;
    }

    // spill P to per-wave LDS tile to re-enter A-fragment layout
#pragma unroll
    for (int r = 0; r < 8; ++r)
#pragma unroll
      for (int n = 0; n < 8; ++n)
        Ps[w][hf * 8 + r][n * 16 + l16] = f2bf(sacc[n][r]);
    __syncthreads();

    // O += P @ V   (A = P 16x128, B^T = VTs)
#pragma unroll
    for (int kk2 = 0; kk2 < 4; ++kk2) {
      const bf16_t* pr = &Ps[w][l16][kk2 * 32];
      v16bf pa = frag_from(pr + (hf ? 16 : 0), pr + (hf ? 24 : 8));
#pragma unroll
      for (int ni = 0; ni < 4; ++ni) {
        const bf16_t* vp = &VTs[ni * 16 + l16][kk2 * 32 + (hf ? 16 : 0)];
        v16bf bfrag = frag_from(vp, vp + 8);
        oacc[ni] = __builtin_amdgcn_wmma_f32_16x16x32_bf16(
            false, pa, false, bfrag, (short)0, oacc[ni], false, false);
      }
    }
    __syncthreads();
  }

  // normalize and write bf16 attention output [T, DIM]
#pragma unroll
  for (int r = 0; r < 8; ++r) {
    float inv = 1.f / lrow[r];
    int tok = seqbase + q0 + w * 16 + hf * 8 + r;
#pragma unroll
    for (int ni = 0; ni < 4; ++ni)
      ob[(size_t)tok * DIM + h * HD + ni * 16 + l16] = f2bf(oacc[ni][r] * inv);
  }
}

// ---------------------------------------------------------------------------
// x2 = x + g_msa * proj ; h2 = bf16( LN(x2)*(1+sc_mlp) + sh_mlp )
// ---------------------------------------------------------------------------
__global__ __launch_bounds__(256)
void k_resid1(const float* __restrict__ x, const float* __restrict__ proj,
              const float* __restrict__ ada, const float* __restrict__ ln2w,
              float* __restrict__ x2, bf16_t* __restrict__ h2) {
  const int t = blockIdx.x, tid = threadIdx.x;
  const int seq = t / L_;
  const float* ab = ada + (size_t)seq * ADAD;
  __shared__ float red[256];
  float vals[4]; float s = 0.f;
#pragma unroll
  for (int i = 0; i < 4; ++i) {
    int dcol = tid + 256 * i;
    float v = x[(size_t)t * DIM + dcol] + ab[2 * DIM + dcol] * proj[(size_t)t * DIM + dcol];
    x2[(size_t)t * DIM + dcol] = v;
    vals[i] = v; s += v;
  }
  red[tid] = s; __syncthreads();
  for (int o = 128; o > 0; o >>= 1) { if (tid < o) red[tid] += red[tid + o]; __syncthreads(); }
  const float mu = red[0] * (1.f / DIM);
  __syncthreads();
  float sq = 0.f;
#pragma unroll
  for (int i = 0; i < 4; ++i) { float d = vals[i] - mu; sq += d * d; }
  red[tid] = sq; __syncthreads();
  for (int o = 128; o > 0; o >>= 1) { if (tid < o) red[tid] += red[tid + o]; __syncthreads(); }
  const float rstd = rsqrtf(red[0] * (1.f / DIM) + EPSV);
#pragma unroll
  for (int i = 0; i < 4; ++i) {
    int dcol = tid + 256 * i;
    float hh = (vals[i] - mu) * rstd * ln2w[dcol];
    hh = hh * (1.f + ab[4 * DIM + dcol]) + ab[3 * DIM + dcol];
    h2[(size_t)t * DIM + dcol] = f2bf(hh);
  }
}

// ---------------------------------------------------------------------------
// out = x2 + g_mlp * (m2 + b2)
// ---------------------------------------------------------------------------
__global__ void k_final(const float* __restrict__ x2, const float* __restrict__ m2,
                        const float* __restrict__ b2, const float* __restrict__ ada,
                        float* __restrict__ out) {
  int idx = blockIdx.x * 256 + threadIdx.x;  // T_*DIM exact
  int t = idx >> 10, dcol = idx & (DIM - 1);
  int seq = t / L_;
  float g = ada[(size_t)seq * ADAD + 5 * DIM + dcol];
  out[idx] = x2[idx] + g * (m2[idx] + b2[dcol]);
}

// ---------------------------------------------------------------------------
extern "C" void kernel_launch(void* const* d_in, const int* in_sizes, int n_in,
                              void* d_out, int out_size, void* d_ws, size_t ws_size,
                              hipStream_t stream) {
  (void)in_sizes; (void)n_in; (void)out_size; (void)ws_size;
  const float* x     = (const float*)d_in[0];
  const float* cosT  = (const float*)d_in[3];
  const float* sinT  = (const float*)d_in[4];
  const long long* pos = (const long long*)d_in[5];
  const float* c     = (const float*)d_in[6];
  const float* ln1w  = (const float*)d_in[7];
  const float* Wq    = (const float*)d_in[8];
  const float* Wk    = (const float*)d_in[9];
  const float* Wv    = (const float*)d_in[10];
  const float* Wo    = (const float*)d_in[11];
  const float* ln2w  = (const float*)d_in[12];
  const float* W1    = (const float*)d_in[13];
  const float* b1    = (const float*)d_in[14];
  const float* W2    = (const float*)d_in[15];
  const float* b2    = (const float*)d_in[16];
  const float* adaw  = (const float*)d_in[17];
  const float* adab  = (const float*)d_in[18];
  float* out = (float*)d_out;

  char* ws = (char*)d_ws;
  const size_t MB = 1ull << 20;
  bf16_t* WqT  = (bf16_t*)(ws + 0 * MB);
  bf16_t* WkT  = (bf16_t*)(ws + 2 * MB);
  bf16_t* WvT  = (bf16_t*)(ws + 4 * MB);
  bf16_t* WoT  = (bf16_t*)(ws + 6 * MB);
  bf16_t* W1T  = (bf16_t*)(ws + 8 * MB);
  bf16_t* W2T  = (bf16_t*)(ws + 16 * MB);
  float*  ada  = (float*) (ws + 24 * MB);
  bf16_t* h1   = (bf16_t*)(ws + 25 * MB);   // 16MB, later reused as attn output
  // region A (41..137 MB)
  float*  qf   = (float*) (ws + 41 * MB);
  float*  kf   = (float*) (ws + 73 * MB);
  float*  vf   = (float*) (ws + 105 * MB);
  float*  m_o  = (float*) (ws + 41 * MB);   // after rope: reuse qf
  bf16_t* g1   = (bf16_t*)(ws + 73 * MB);   // after rope: reuse kf/vf (64MB)
  float*  m2   = (float*) (ws + 41 * MB);   // after resid1: reuse m_o
  // region B (137..185 MB)
  bf16_t* qbuf = (bf16_t*)(ws + 137 * MB);
  bf16_t* kbuf = (bf16_t*)(ws + 153 * MB);
  bf16_t* vbuf = (bf16_t*)(ws + 169 * MB);
  float*  x2   = (float*) (ws + 137 * MB);  // after attention: reuse qbuf/kbuf
  bf16_t* h2   = (bf16_t*)(ws + 169 * MB);  // after attention: reuse vbuf
  bf16_t* attnb = h1;

  // 1) weight prep
  k_transpose_bf16<<<(DIM * DIM) / 256, 256, 0, stream>>>(Wq, WqT, DIM, DIM);
  k_transpose_bf16<<<(DIM * DIM) / 256, 256, 0, stream>>>(Wk, WkT, DIM, DIM);
  k_transpose_bf16<<<(DIM * DIM) / 256, 256, 0, stream>>>(Wv, WvT, DIM, DIM);
  k_transpose_bf16<<<(DIM * DIM) / 256, 256, 0, stream>>>(Wo, WoT, DIM, DIM);
  k_transpose_bf16<<<(DIM * MLPD) / 256, 256, 0, stream>>>(W1, W1T, DIM, MLPD);
  k_transpose_bf16<<<(MLPD * DIM) / 256, 256, 0, stream>>>(W2, W2T, MLPD, DIM);
  // 2) adaLN modulation
  k_ada<<<(B_ * ADAD) / 256, 256, 0, stream>>>(c, adaw, adab, ada);
  // 3) h1 = LN(x)*(1+sc_msa)+sh_msa
  k_ln_mod<<<T_, 256, 0, stream>>>(x, ln1w, ada, 0 * DIM, 1 * DIM, h1);
  // 4) QKV projections
  k_gemm<0><<<dim3(T_ / BM, DIM / BN), 256, 0, stream>>>(T_, DIM, DIM, h1, WqT, qf, nullptr, nullptr);
  k_gemm<0><<<dim3(T_ / BM, DIM / BN), 256, 0, stream>>>(T_, DIM, DIM, h1, WkT, kf, nullptr, nullptr);
  k_gemm<0><<<dim3(T_ / BM, DIM / BN), 256, 0, stream>>>(T_, DIM, DIM, h1, WvT, vf, nullptr, nullptr);
  // 5) RoPE + bf16 convert
  k_rope_cvt<<<(T_ * NH * 32) / 256, 256, 0, stream>>>(qf, kf, vf, cosT, sinT, pos, qbuf, kbuf, vbuf);
  // 6) attention
  k_attn<<<dim3(B_ * NH, L_ / 128), 256, 0, stream>>>(qbuf, kbuf, vbuf, attnb);
  // 7) O projection
  k_gemm<0><<<dim3(T_ / BM, DIM / BN), 256, 0, stream>>>(T_, DIM, DIM, attnb, WoT, m_o, nullptr, nullptr);
  // 8) residual + LN2 modulation
  k_resid1<<<T_, 256, 0, stream>>>(x, m_o, ada, ln2w, x2, h2);
  // 9) MLP up-proj with fused bias+gelu -> bf16
  k_gemm<2><<<dim3(T_ / BM, MLPD / BN), 256, 0, stream>>>(T_, MLPD, DIM, h2, W1T, nullptr, g1, b1);
  // 10) MLP down-proj
  k_gemm<0><<<dim3(T_ / BM, DIM / BN), 256, 0, stream>>>(T_, DIM, MLPD, g1, W2T, m2, nullptr, nullptr);
  // 11) final residual
  k_final<<<(T_ * DIM) / 256, 256, 0, stream>>>(x2, m2, b2, ada, out);
}
